// LlamaSdpaAttention_730144440681
// MI455X (gfx1250) — matrix operand order repaired and checked by reading, not compiled, not run on hardware
//
#include <hip/hip_runtime.h>
#include <hip/hip_bf16.h>

// ---------------------------------------------------------------------------
// LlamaSdpaAttention on MI455X (gfx1250): bf16 WMMA GEMMs + flash attention.
// B=1, T=2048, C=3072, H=24, HKV=8, D=128, G=3
// ---------------------------------------------------------------------------

#define T_   2048
#define C_   3072
#define H_   24
#define HKV_ 8
#define D_   128
#define G_   3

typedef __attribute__((ext_vector_type(16))) __bf16   v16bf;
typedef __attribute__((ext_vector_type(8)))  float    v8f;
typedef __attribute__((ext_vector_type(4)))  unsigned uv4;
typedef __attribute__((ext_vector_type(4)))  int      v4i;

union frag16 { v16bf v; uv4 u[2]; };

#define HAS_ASYNC_LDS __has_builtin(__builtin_amdgcn_global_load_async_to_lds_b128)

typedef __attribute__((address_space(1))) v4i* gv4i_p;   // global int4*
typedef __attribute__((address_space(3))) v4i* sv4i_p;   // LDS int4*
__device__ __forceinline__ gv4i_p as1v(const void* p) {
    return (gv4i_p)(unsigned long long)p;
}
__device__ __forceinline__ sv4i_p as3v(void* p) {
    return (sv4i_p)(unsigned int)(unsigned long long)p;  // LDS offset = low 32 bits
}
__device__ __forceinline__ void wait_async_le4() {
#if __has_builtin(__builtin_amdgcn_s_wait_asynccnt)
    __builtin_amdgcn_s_wait_asynccnt(4);
#else
    asm volatile("s_wait_asynccnt 0x4" ::: "memory");
#endif
}
__device__ __forceinline__ void wait_async_le0() {
#if __has_builtin(__builtin_amdgcn_s_wait_asynccnt)
    __builtin_amdgcn_s_wait_asynccnt(0);
#else
    asm volatile("s_wait_asynccnt 0x0" ::: "memory");
#endif
}

__device__ inline unsigned short f2bf(float f) {
    union { float f; unsigned u; } cv; cv.f = f;
    unsigned u = cv.u;
    u += 0x7FFFu + ((u >> 16) & 1u);          // round-to-nearest-even
    return (unsigned short)(u >> 16);
}
__device__ inline float bf2f(unsigned short b) {
    union { unsigned u; float f; } cv; cv.u = ((unsigned)b) << 16;
    return cv.f;
}
__device__ inline v8f vzero8() {
    v8f z;
    #pragma unroll
    for (int i = 0; i < 8; ++i) z[i] = 0.0f;
    return z;
}

__device__ inline float rmax16(float v) {
    #pragma unroll
    for (int m = 1; m < 16; m <<= 1) v = fmaxf(v, __shfl_xor(v, m, 32));
    return v;
}
__device__ inline float rsum16(float v) {
    #pragma unroll
    for (int m = 1; m < 16; m <<= 1) v += __shfl_xor(v, m, 32);
    return v;
}

// ---------------------------------------------------------------------------
// fp32 -> bf16 elementwise convert
// ---------------------------------------------------------------------------
__global__ __launch_bounds__(256)
void cvt_f32_to_bf16(const float* __restrict__ in, unsigned short* __restrict__ out, int n) {
    int i = blockIdx.x * 256 + threadIdx.x;
    if (i < n) out[i] = f2bf(in[i]);
}

// ---------------------------------------------------------------------------
// Tiled transpose + convert: W [K][N] fp32 -> Wt [N][K] bf16.
// 64x64 tile through LDS: coalesced float4 reads, coalesced 32B writes.
// ---------------------------------------------------------------------------
__global__ __launch_bounds__(256)
void transpose_to_bf16(const float* __restrict__ W, unsigned short* __restrict__ Wt,
                       int Kdim, int Ndim) {
    __shared__ unsigned short tile[64][68];           // pad to dodge bank conflicts
    const int k0 = blockIdx.x * 64;                   // row block of W
    const int n0 = blockIdx.y * 64;                   // col block of W
    const int t  = threadIdx.x;

    const int tx = t & 15;                            // j = tx*4
    const int ty = t >> 4;
    #pragma unroll
    for (int p = 0; p < 4; ++p) {
        const int i = ty + p * 16;
        float4 f = *(const float4*)(W + (size_t)(k0 + i) * Ndim + n0 + tx * 4);
        unsigned short* d = &tile[i][tx * 4];
        d[0] = f2bf(f.x); d[1] = f2bf(f.y); d[2] = f2bf(f.z); d[3] = f2bf(f.w);
    }
    __syncthreads();

    const int r    = t >> 2;                          // output row (n)
    const int koff = (t & 3) * 16;                    // 16 k-elements per thread
    unsigned short buf[16];
    #pragma unroll
    for (int q = 0; q < 16; ++q) buf[q] = tile[koff + q][r];
    uv4* dst = (uv4*)(Wt + (size_t)(n0 + r) * Kdim + k0 + koff);
    dst[0] = *(uv4*)&buf[0];
    dst[1] = *(uv4*)&buf[8];
}

// ---------------------------------------------------------------------------
// Tiled bf16 WMMA GEMM:  Out[M][N] = A[M][K] * Bt[N][K]^T
// 256 threads = 8 waves (4 M x 2 N), wave tile 32x64, block tile 128x128,
// K-step 32.  A/B tiles double-buffered in LDS; filled with the CDNA5
// async global->LDS path (ASYNCcnt) when available, else load+ds_store.
// LAYOUT 0: fp32 out [M][N]
// LAYOUT 1: bf16 out [col>>7][M][128]     (head-major Q/K)
// LAYOUT 2: bf16 out [col>>7][128][M]     (head-major transposed V)
// ---------------------------------------------------------------------------
#define SK 40   // padded LDS row stride (elements): 80B = 20 banks, conflict-free

template <int LAYOUT>
__global__ __launch_bounds__(256)
void gemm_bf16_wmma(const unsigned short* __restrict__ A,
                    const unsigned short* __restrict__ Bt,
                    void* __restrict__ outp, int M, int N, int K) {
    __shared__ unsigned short Asm[2][128 * SK];
    __shared__ unsigned short Bsm[2][128 * SK];

    const int tid     = threadIdx.x;
    const int lane    = tid & 31;
    const int wave    = tid >> 5;           // 0..7
    const int wm      = wave >> 1;          // 0..3
    const int wn      = wave & 1;           // 0..1
    const int half_id = lane >> 4;
    const int ln      = lane & 15;

    const int gm0 = blockIdx.x * 128;
    const int gn0 = blockIdx.y * 128;

    v8f acc[2][4];
    #pragma unroll
    for (int i = 0; i < 2; ++i)
        #pragma unroll
        for (int j = 0; j < 4; ++j) acc[i][j] = vzero8();

    // cooperative staging: each thread moves 16 A-elems + 16 B-elems per tile
    const int srow = tid >> 1;              // 0..127
    const int scol = (tid & 1) * 16;        // 0 or 16

    auto stage = [&](int buf, int k0) {
        const unsigned short* ga = A  + (size_t)(gm0 + srow) * K + k0 + scol;
        const unsigned short* gb = Bt + (size_t)(gn0 + srow) * K + k0 + scol;
        unsigned short* la = &Asm[buf][srow * SK + scol];
        unsigned short* lb = &Bsm[buf][srow * SK + scol];
#if HAS_ASYNC_LDS
        __builtin_amdgcn_global_load_async_to_lds_b128(as1v(ga),     as3v(la),     0, 0);
        __builtin_amdgcn_global_load_async_to_lds_b128(as1v(ga + 8), as3v(la + 8), 0, 0);
        __builtin_amdgcn_global_load_async_to_lds_b128(as1v(gb),     as3v(lb),     0, 0);
        __builtin_amdgcn_global_load_async_to_lds_b128(as1v(gb + 8), as3v(lb + 8), 0, 0);
#else
        uv4 a0 = *(const uv4*)ga, a1 = *(const uv4*)(ga + 8);
        uv4 b0 = *(const uv4*)gb, b1 = *(const uv4*)(gb + 8);
        *(uv4*)la = a0; *(uv4*)(la + 8) = a1;
        *(uv4*)lb = b0; *(uv4*)(lb + 8) = b1;
#endif
    };

    auto compute = [&](int buf) {
        frag16 af[2];
        #pragma unroll
        for (int i = 0; i < 2; ++i) {
            // A layout: elem j<8 -> k = half*8 + j ; j>=8 -> k = 16 + half*8 + (j-8)
            const unsigned short* ap = &Asm[buf][(wm * 32 + i * 16 + ln) * SK];
            af[i].u[0] = *(const uv4*)(ap + half_id * 8);
            af[i].u[1] = *(const uv4*)(ap + 16 + half_id * 8);
        }
        #pragma unroll
        for (int j = 0; j < 4; ++j) {
            // B layout: elem j<8 -> k = half*8 + j ; j>=8 -> k = 16 + half*8 + (j-8)
            const unsigned short* bp = &Bsm[buf][(wn * 64 + j * 16 + ln) * SK];
            frag16 bf;
            bf.u[0] = *(const uv4*)(bp + half_id * 8);
            bf.u[1] = *(const uv4*)(bp + 16 + half_id * 8);
            #pragma unroll
            for (int i = 0; i < 2; ++i)
                acc[i][j] = __builtin_amdgcn_wmma_f32_16x16x32_bf16(
                    false, af[i].v, false, bf.v, (short)0, acc[i][j], false, false);
        }
    };

#if HAS_ASYNC_LDS
    stage(0, 0);
    for (int k0 = 0; k0 < K; k0 += 32) {
        const int cur = (k0 >> 5) & 1;
        if (k0 + 32 < K) {
            stage(1 - cur, k0 + 32);
            wait_async_le4();               // current tile's 4 async ops done
        } else {
            wait_async_le0();
        }
        __syncthreads();
        compute(cur);
        __syncthreads();                    // readers done before next overwrite
    }
#else
    for (int k0 = 0; k0 < K; k0 += 32) {
        const int cur = (k0 >> 5) & 1;
        stage(cur, k0);
        __syncthreads();
        compute(cur);
    }
#endif

    // C/D layout: row = 8*half + r, col = lane%16
    #pragma unroll
    for (int i = 0; i < 2; ++i) {
        #pragma unroll
        for (int j = 0; j < 4; ++j) {
            const int col = gn0 + wn * 64 + j * 16 + ln;
            #pragma unroll
            for (int r = 0; r < 8; ++r) {
                const int row = gm0 + wm * 32 + i * 16 + half_id * 8 + r;
                const float val = acc[i][j][r];
                if (LAYOUT == 0) {
                    ((float*)outp)[(size_t)row * N + col] = val;
                } else if (LAYOUT == 1) {
                    const int h = col >> 7, d = col & 127;
                    ((unsigned short*)outp)[((size_t)h * M + row) * 128 + d] = f2bf(val);
                } else {
                    const int h = col >> 7, d = col & 127;
                    ((unsigned short*)outp)[((size_t)h * 128 + d) * M + row] = f2bf(val);
                }
            }
        }
    }
}

// ---------------------------------------------------------------------------
// RoPE in-place on Qh [H][T][128] and Kh [HKV][T][128] (bf16)
// q'[j] = q[j]*cos - q[j+64]*sin ; q'[j+64] = q[j+64]*cos + q[j]*sin
// ---------------------------------------------------------------------------
__global__ __launch_bounds__(256)
void rope_inplace(unsigned short* __restrict__ Qh, unsigned short* __restrict__ Kh) {
    int idx = blockIdx.x * 256 + threadIdx.x;       // (H+HKV)*T*64 threads
    int j    = idx & 63;
    int t    = (idx >> 6) & (T_ - 1);
    int head = idx >> 17;                           // 64*2048 = 2^17
    if (head >= H_ + HKV_) return;
    float inv = __powf(10000.0f, -(float)(2 * j) * (1.0f / 128.0f));
    float f = (float)t * inv;
    float s, c;
    __sincosf(f, &s, &c);
    unsigned short* base = (head < H_)
        ? (Qh + ((size_t)head * T_ + t) * 128)
        : (Kh + ((size_t)(head - H_) * T_ + t) * 128);
    float x0 = bf2f(base[j]), x1 = bf2f(base[j + 64]);
    base[j]      = f2bf(x0 * c - x1 * s);
    base[j + 64] = f2bf(x1 * c + x0 * s);
}

// ---------------------------------------------------------------------------
// Causal flash attention, one wave (32 threads) per (head, 16-row q-block).
// Qh [H][T][128], Kh [HKV][T][128], Vt [HKV][128][T]  (all bf16)
// Output Ab [T][C] bf16 (col = h*128 + d) — A-matrix of the final GEMM.
// ---------------------------------------------------------------------------
__global__ __launch_bounds__(32)
void attn_causal_wmma(const unsigned short* __restrict__ Qh,
                      const unsigned short* __restrict__ Kh,
                      const unsigned short* __restrict__ Vt,
                      unsigned short* __restrict__ Ab) {
    __shared__ unsigned short Plds[16 * 32];

    const int qb      = blockIdx.x * 16;
    const int h       = blockIdx.y;
    const int hkv     = h / G_;
    const int lane    = threadIdx.x;
    const int half_id = lane >> 4;
    const int ln      = lane & 15;

    const unsigned short* Qp = Qh + (size_t)h   * T_ * 128;
    const unsigned short* Kp = Kh + (size_t)hkv * T_ * 128;
    const unsigned short* Vp = Vt + (size_t)hkv * 128 * T_;

    // Q A-fragments over the full d=128 (4 x k-step 32), kept in registers
    frag16 qf[4];
    const size_t qrow = (size_t)(qb + ln) * 128;
    #pragma unroll
    for (int kk = 0; kk < 4; ++kk) {
        qf[kk].u[0] = *(const uv4*)(Qp + qrow + kk * 32 + half_id * 8);
        qf[kk].u[1] = *(const uv4*)(Qp + qrow + kk * 32 + 16 + half_id * 8);
    }

    float mi[8], li[8];
    v8f O[8];
    #pragma unroll
    for (int r = 0; r < 8; ++r) { mi[r] = -__builtin_inff(); li[r] = 0.0f; }
    #pragma unroll
    for (int db = 0; db < 8; ++db) O[db] = vzero8();

    const float scale = 0.08838834764831845f;   // 1/sqrt(128)
    const float L2E   = 1.4426950408889634f;

    const int ktiles = (qb + 47) / 32;          // cover keys 0..qb+15
    for (int tile = 0; tile < ktiles; ++tile) {
        const int kb = tile * 32;

        // S = Q * K^T : two 16x16 column blocks, 4 k-steps each
        v8f S0 = vzero8(), S1 = vzero8();
        #pragma unroll
        for (int kk = 0; kk < 4; ++kk) {
            frag16 kf0, kf1;
            const unsigned short* kp0 = Kp + (size_t)(kb + ln) * 128 + kk * 32 + half_id * 8;
            const unsigned short* kp1 = Kp + (size_t)(kb + 16 + ln) * 128 + kk * 32 + half_id * 8;
            kf0.u[0] = *(const uv4*)kp0; kf0.u[1] = *(const uv4*)(kp0 + 16);
            kf1.u[0] = *(const uv4*)kp1; kf1.u[1] = *(const uv4*)(kp1 + 16);
            S0 = __builtin_amdgcn_wmma_f32_16x16x32_bf16(false, qf[kk].v, false, kf0.v,
                                                         (short)0, S0, false, false);
            S1 = __builtin_amdgcn_wmma_f32_16x16x32_bf16(false, qf[kk].v, false, kf1.v,
                                                         (short)0, S1, false, false);
        }

        // online softmax (fp32), per-lane rows: row = qb + 8*half + r, key = kb(+16) + ln
        const int key0 = kb + ln, key1 = kb + 16 + ln;
        #pragma unroll
        for (int r = 0; r < 8; ++r) {
            const int row = qb + half_id * 8 + r;
            float s0 = (key0 <= row) ? S0[r] * scale : -__builtin_inff();
            float s1 = (key1 <= row) ? S1[r] * scale : -__builtin_inff();
            float tmax = rmax16(fmaxf(s0, s1));
            float mn = fmaxf(mi[r], tmax);
            float alpha = exp2f((mi[r] - mn) * L2E);
            float p0 = exp2f((s0 - mn) * L2E);
            float p1 = exp2f((s1 - mn) * L2E);
            float rs = rsum16(p0 + p1);
            li[r] = li[r] * alpha + rs;
            mi[r] = mn;
            #pragma unroll
            for (int db = 0; db < 8; ++db) O[db][r] *= alpha;
            Plds[(half_id * 8 + r) * 32 + ln]      = f2bf(p0);
            Plds[(half_id * 8 + r) * 32 + 16 + ln] = f2bf(p1);
        }
        __syncthreads();

        // reload P as an A-fragment (16x32)
        frag16 pf;
        pf.u[0] = *(const uv4*)(&Plds[ln * 32 + half_id * 8]);
        pf.u[1] = *(const uv4*)(&Plds[ln * 32 + 16 + half_id * 8]);
        __syncthreads();

        // O += P * V : 8 d-blocks, B-fragments contiguous in key from Vt
        #pragma unroll
        for (int db = 0; db < 8; ++db) {
            frag16 vf;
            const unsigned short* vp = Vp + (size_t)(db * 16 + ln) * T_ + kb + half_id * 8;
            vf.u[0] = *(const uv4*)vp; vf.u[1] = *(const uv4*)(vp + 16);
            O[db] = __builtin_amdgcn_wmma_f32_16x16x32_bf16(false, pf.v, false, vf.v,
                                                            (short)0, O[db], false, false);
        }
    }

    // epilogue: normalize and scatter to Ab [T][C] bf16, col = h*128 + db*16 + ln
    #pragma unroll
    for (int r = 0; r < 8; ++r) {
        const float invl = 1.0f / li[r];
        const int row = qb + half_id * 8 + r;
        #pragma unroll
        for (int db = 0; db < 8; ++db) {
            const int col = h * 128 + db * 16 + ln;
            Ab[(size_t)row * C_ + col] = f2bf(O[db][r] * invl);
        }
    }
}

// ---------------------------------------------------------------------------
// launch
// ---------------------------------------------------------------------------
extern "C" void kernel_launch(void* const* d_in, const int* in_sizes, int n_in,
                              void* d_out, int out_size, void* d_ws, size_t ws_size,
                              hipStream_t stream) {
    const float* x  = (const float*)d_in[0];   // [T][C]
    const float* Wq = (const float*)d_in[1];   // [C][3072]
    const float* Wk = (const float*)d_in[2];   // [C][1024]
    const float* Wv = (const float*)d_in[3];   // [C][1024]
    const float* Wo = (const float*)d_in[4];   // [C][3072]
    float* out = (float*)d_out;                // [T][C]
    (void)in_sizes; (void)n_in; (void)out_size; (void)ws_size;

    // workspace layout (bf16 = ushort), byte offsets
    char* ws = (char*)d_ws;
    unsigned short* xb  = (unsigned short*)(ws);                         // T*C
    unsigned short* WqT = (unsigned short*)(ws + 12582912);              // 3072*3072
    unsigned short* WkT = (unsigned short*)(ws + 31457280);              // 1024*3072
    unsigned short* WvT = (unsigned short*)(ws + 37748736);              // 1024*3072
    unsigned short* WoT = (unsigned short*)(ws + 44040192);              // 3072*3072
    unsigned short* Qh  = (unsigned short*)(ws + 62914560);              // H*T*128
    unsigned short* Kh  = (unsigned short*)(ws + 75497472);              // HKV*T*128
    unsigned short* Vt  = (unsigned short*)(ws + 79691776);              // HKV*128*T
    unsigned short* Ab  = (unsigned short*)(ws + 83886080);              // T*C

    // 1) convert x, tiled transpose+convert weights
    cvt_f32_to_bf16<<<(T_ * C_) / 256, 256, 0, stream>>>(x, xb, T_ * C_);
    transpose_to_bf16<<<dim3(C_ / 64, 3072 / 64), 256, 0, stream>>>(Wq, WqT, C_, 3072);
    transpose_to_bf16<<<dim3(C_ / 64, 1024 / 64), 256, 0, stream>>>(Wk, WkT, C_, 1024);
    transpose_to_bf16<<<dim3(C_ / 64, 1024 / 64), 256, 0, stream>>>(Wv, WvT, C_, 1024);
    transpose_to_bf16<<<dim3(C_ / 64, 3072 / 64), 256, 0, stream>>>(Wo, WoT, C_, 3072);

    // 2) QKV projections (block tile 128x128, LDS-staged)
    gemm_bf16_wmma<1><<<dim3(T_ / 128, 3072 / 128), 256, 0, stream>>>(xb, WqT, Qh, T_, 3072, C_);
    gemm_bf16_wmma<1><<<dim3(T_ / 128, 1024 / 128), 256, 0, stream>>>(xb, WkT, Kh, T_, 1024, C_);
    gemm_bf16_wmma<2><<<dim3(T_ / 128, 1024 / 128), 256, 0, stream>>>(xb, WvT, Vt, T_, 1024, C_);

    // 3) RoPE on Q and K
    rope_inplace<<<((H_ + HKV_) * T_ * 64) / 256, 256, 0, stream>>>(Qh, Kh);

    // 4) causal flash attention -> Ab [T][C] bf16
    attn_causal_wmma<<<dim3(T_ / 16, H_), 32, 0, stream>>>(Qh, Kh, Vt, Ab);

    // 5) output projection -> fp32 d_out
    gemm_bf16_wmma<0><<<dim3(T_ / 128, 3072 / 128), 256, 0, stream>>>(Ab, WoT, out, T_, 3072, C_);
}
